// Gaussian_65034394796366
// MI455X (gfx1250) — matrix operand order repaired and checked by reading, not compiled, not run on hardware
//
#include <hip/hip_runtime.h>

// ---------------------------------------------------------------------------
// Gaussian kernel  out[n,m] = exp(-0.5 * sum_d (x[n,d]-mu[m,d])^2 / cov[m,d])
// N=65536, M=1024, D=256  ->  two N*M*D GEMMs + bias + exp.
//
// MI455X plan:
//  * bf16 hi/lo split GEMM (3 WMMA products per term, lo*lo dropped):
//    ~2^-16 relative accuracy at V_WMMA_F32_16X16X32_BF16 rate.
//  * PACKED operand layouts: per (row, k-chunk-of-32) a 256B record holding
//    all 4 bf16 variants -> one base VGPR + immediate offsets for every load.
//  * B operands staged to LDS with GLOBAL_LOAD_ASYNC_TO_LDS_B128
//    (double-buffered, s_wait_asynccnt), killing the 8x redundant L2 reads.
//  * Each wave computes 2 N-strips (32 rows) so every B fragment read from
//    LDS feeds 12 WMMAs -> LDS bandwidth below WMMA demand.
//  * exp via v_exp_f32; 268MB output streamed with non-temporal stores.
//
// Workspace (assumed ws_size >= ~131 MB):
//   [0)       Apk : N * 8ktiles * {sq_hi,sq_lo,x_hi,x_lo} * 32 bf16 = 128 MB
//   [+128MB)  Bpk : M * 8ktiles * {ic_hi,ic_lo,b2_hi,b2_lo} * 32 bf16 = 2 MB
//   [+130MB)  term_m : M f32
// ---------------------------------------------------------------------------

#define N_PTS 65536
#define M_CL  1024
#define D_DIM 256

typedef __attribute__((ext_vector_type(16))) __bf16 v16bf;
typedef __attribute__((ext_vector_type(8)))  float  v8f;

union FragU {
    uint4 u[2];   // 32 bytes
    v16bf v;      // 16 bf16
};

__device__ __forceinline__ unsigned short f2bf_rne(float f) {
    unsigned u = __builtin_bit_cast(unsigned, f);
    u += 0x7FFFu + ((u >> 16) & 1u);
    return (unsigned short)(u >> 16);
}
__device__ __forceinline__ float bf2f(unsigned short h) {
    return __builtin_bit_cast(float, (unsigned)h << 16);
}

// ---------------------------------------------------------------------------
// Prep kernels: write PACKED layouts.
// Record for row r, k-chunk kt (32 d's):  elem offset = r*1024 + kt*128
//   [ +0 ) hi of primary   (x^2 | 1/cov)      32 bf16
//   [ +32) lo of primary                      32 bf16
//   [ +64) hi of secondary (x   | -2*mu/cov)  32 bf16
//   [ +96) lo of secondary                    32 bf16
// ---------------------------------------------------------------------------
__global__ void prep_x_kernel(const float* __restrict__ x,
                              unsigned short* __restrict__ Apk) {
    int idx = blockIdx.x * blockDim.x + threadIdx.x;   // 0..N*D-1
    int n = idx >> 8, d = idx & 255;
    int kt = d >> 5, kk = d & 31;
    unsigned short* p = Apk + (size_t)n * 1024 + kt * 128 + kk;
    float v = x[idx];
    float s = v * v;
    unsigned short h;
    h = f2bf_rne(s); p[0]  = h; p[32] = f2bf_rne(s - bf2f(h));
    h = f2bf_rne(v); p[64] = h; p[96] = f2bf_rne(v - bf2f(h));
}

__global__ void prep_b_kernel(const float* __restrict__ mu,
                              const float* __restrict__ cov,
                              unsigned short* __restrict__ Bpk) {
    int idx = blockIdx.x * blockDim.x + threadIdx.x;   // 0..M*D-1
    int m = idx >> 8, d = idx & 255;
    int kt = d >> 5, kk = d & 31;
    unsigned short* p = Bpk + (size_t)m * 1024 + kt * 128 + kk;
    float ic = 1.0f / cov[idx];
    float b2 = -2.0f * mu[idx] * ic;
    unsigned short h;
    h = f2bf_rne(ic); p[0]  = h; p[32] = f2bf_rne(ic - bf2f(h));
    h = f2bf_rne(b2); p[64] = h; p[96] = f2bf_rne(b2 - bf2f(h));
}

__global__ void prep_term_kernel(const float* __restrict__ mu,
                                 const float* __restrict__ cov,
                                 float* __restrict__ term) {
    int m = blockIdx.x * blockDim.x + threadIdx.x;     // 0..M-1
    float s = 0.0f;
    const float* mr = mu  + (size_t)m * D_DIM;
    const float* cr = cov + (size_t)m * D_DIM;
    for (int d = 0; d < D_DIM; ++d) {
        float mv = mr[d];
        s += mv * mv / cr[d];
    }
    term[m] = s;
}

// ---------------------------------------------------------------------------
// Async copy: 16B per lane per issue; IOFFSET applies to BOTH the global
// source and the LDS destination (ISA 07_vmem 15.18.3), so the chunk offset
// lives entirely in the immediate.
// ---------------------------------------------------------------------------
template <int OFF>
__device__ __forceinline__ void async_cp16(unsigned lds_off, unsigned goff,
                                           const unsigned short* __restrict__ base) {
    asm volatile("global_load_async_to_lds_b128 %0, %1, %2 offset:%3"
                 :: "v"(lds_off), "v"(goff), "s"(base), "n"(OFF)
                 : "memory");
}

#define WMMA_BF16(A, B, C) \
    __builtin_amdgcn_wmma_f32_16x16x32_bf16(false, (A), false, (B), (short)0, (C), false, false)

// ---------------------------------------------------------------------------
// Main kernel: 256 threads = 8 wave32.  Block tile = 256 N rows x 128 M cols.
//   blockIdx.x : N/256 = 256 tiles;  blockIdx.y : M/128 = 8 chunks.
// Each wave: 2 strips of 16 rows, 8 M tiles -> 16 accumulator tiles (128 VGPR).
// K loop: 8 chunks of 32; B staged in LDS (2 x 32KB double buffer).
// ---------------------------------------------------------------------------
__global__ __launch_bounds__(256)
void gauss_wmma_kernel(const unsigned short* __restrict__ Apk,
                       const unsigned short* __restrict__ Bpk,
                       const float* __restrict__ term,
                       float* __restrict__ out) {
    extern __shared__ char smem[];   // 64 KB

    const int tid   = threadIdx.x;
    const int lane  = tid & 31;
    const int row16 = lane & 15;
    const int kg    = lane >> 4;
    const int wave  = tid >> 5;
    const int n0    = blockIdx.x * 256 + wave * 32;   // wave's first row
    const int mbase = blockIdx.y * 128;               // chunk's first col

    // A: per-thread base; strip/kt/variant/half all compile-time immediates.
    const char* abase = (const char*)Apk + (size_t)(n0 + row16) * 2048 + kg * 32;

    // B staging: thread (msel,half) copies 128B of m-row (mbase+msel)'s chunk.
    const int msel = tid >> 1;                        // 0..127
    const int half = tid & 1;
    const unsigned goff0   = (unsigned)((mbase + msel) * 2048 + half * 128);
    const unsigned ldsoff0 = (unsigned)(msel * 256 + half * 128);

    // LDS fragment read base (per lane); everything else is immediate.
    const unsigned frag0 = (unsigned)(row16 * 256 + kg * 32);

    v8f acc[2][8] = {};

#define STAGE(KT)                                                        \
    {                                                                    \
        unsigned g = goff0 + (unsigned)((KT) * 256);                     \
        unsigned l = ldsoff0 + (unsigned)(((KT) & 1) * 32768);           \
        async_cp16<0>(l, g, Bpk);   async_cp16<16>(l, g, Bpk);           \
        async_cp16<32>(l, g, Bpk);  async_cp16<48>(l, g, Bpk);           \
        async_cp16<64>(l, g, Bpk);  async_cp16<80>(l, g, Bpk);           \
        async_cp16<96>(l, g, Bpk);  async_cp16<112>(l, g, Bpk);          \
    }

    STAGE(0)

#pragma unroll
    for (int kt = 0; kt < D_DIM / 32; ++kt) {
        if (kt < 7) {
            STAGE(kt + 1)
            asm volatile("s_wait_asynccnt 8" ::: "memory");  // stage kt done
        } else {
            asm volatile("s_wait_asynccnt 0" ::: "memory");
        }
        __syncthreads();   // publish stage kt to all waves

        // A fragments for both strips: {sq_hi, sq_lo, x_hi, x_lo}
        v16bf a[2][4];
#pragma unroll
        for (int s = 0; s < 2; ++s)
#pragma unroll
            for (int v = 0; v < 4; ++v) {
                const uint4* p =
                    (const uint4*)(abase + s * 32768 + kt * 256 + v * 64);
                FragU f;
                f.u[0] = p[0];
                f.u[1] = p[1];   // +16B (K = kg*16+8 .. +15)
                a[s][v] = f.v;
            }

        const char* bbuf = smem + ((kt & 1) * 32768);
#pragma unroll
        for (int mt = 0; mt < 8; ++mt) {
            const char* bp = bbuf + (frag0 + mt * 4096);
            FragU f;
            const uint4* q;
            q = (const uint4*)(bp +   0); f.u[0]=q[0]; f.u[1]=q[1]; v16bf b1h = f.v;
            q = (const uint4*)(bp +  64); f.u[0]=q[0]; f.u[1]=q[1]; v16bf b1l = f.v;
            q = (const uint4*)(bp + 128); f.u[0]=q[0]; f.u[1]=q[1]; v16bf b2h = f.v;
            q = (const uint4*)(bp + 192); f.u[0]=q[0]; f.u[1]=q[1]; v16bf b2l = f.v;
#pragma unroll
            for (int s = 0; s < 2; ++s) {
                v8f c = acc[s][mt];
                c = WMMA_BF16(a[s][0], b1h, c);   // x^2hi * ic_hi
                c = WMMA_BF16(a[s][0], b1l, c);   // x^2hi * ic_lo
                c = WMMA_BF16(a[s][1], b1h, c);   // x^2lo * ic_hi
                c = WMMA_BF16(a[s][2], b2h, c);   // x_hi  * b2_hi
                c = WMMA_BF16(a[s][2], b2l, c);   // x_hi  * b2_lo
                c = WMMA_BF16(a[s][3], b2h, c);   // x_lo  * b2_hi
                acc[s][mt] = c;
            }
        }
        __syncthreads();   // all reads of this buffer done before reuse
    }

    // Epilogue: quad = acc + term_m[col]; out = exp2(-0.5*log2e * quad)
    // C/D layout: vgpr v, lane l -> row = v + 8*kg, col = l&15.
    const float kc = -0.72134752044448169f;
#pragma unroll
    for (int mt = 0; mt < 8; ++mt) {
        const int col = mbase + mt * 16 + row16;
        const float t = term[col];
#pragma unroll
        for (int s = 0; s < 2; ++s)
#pragma unroll
            for (int v = 0; v < 8; ++v) {
                const int row = n0 + s * 16 + v + 8 * kg;
                float q = acc[s][mt][v] + t;
                float r = __builtin_amdgcn_exp2f(kc * q);
                __builtin_nontemporal_store(r, &out[(size_t)row * M_CL + col]);
            }
    }
#undef STAGE
}

// ---------------------------------------------------------------------------
extern "C" void kernel_launch(void* const* d_in, const int* in_sizes, int n_in,
                              void* d_out, int out_size, void* d_ws, size_t ws_size,
                              hipStream_t stream) {
    const float* x   = (const float*)d_in[0];   // N x D
    const float* mu  = (const float*)d_in[1];   // M x 1 x D
    const float* cov = (const float*)d_in[2];   // M x D
    float* out = (float*)d_out;                 // N x M

    char* ws = (char*)d_ws;
    const size_t A_BYTES = (size_t)N_PTS * 1024 * sizeof(unsigned short); // 128 MB
    const size_t B_BYTES = (size_t)M_CL  * 1024 * sizeof(unsigned short); //   2 MB

    unsigned short* Apk  = (unsigned short*)(ws);
    unsigned short* Bpk  = (unsigned short*)(ws + A_BYTES);
    float*          termP = (float*)(ws + A_BYTES + B_BYTES);

    prep_x_kernel<<<(N_PTS * D_DIM) / 256, 256, 0, stream>>>(x, Apk);
    prep_b_kernel<<<(M_CL * D_DIM) / 256, 256, 0, stream>>>(mu, cov, Bpk);
    prep_term_kernel<<<M_CL / 256, 256, 0, stream>>>(mu, cov, termP);

    dim3 grid(N_PTS / 256, M_CL / 128);   // 256 x 8
    gauss_wmma_kernel<<<grid, 256, 65536, stream>>>(Apk, Bpk, termP, out);
}
